// RoutedFFN_5738076307889
// MI455X (gfx1250) — compile-verified, matrix operand order checked by reading, not run on hardware
//
#include <hip/hip_runtime.h>
#include <hip/hip_bf16.h>
#include <cfloat>

typedef __attribute__((ext_vector_type(16))) __bf16 bf16x16;
typedef __attribute__((ext_vector_type(8)))  __bf16 bf16x8;
typedef __attribute__((ext_vector_type(8)))  float  v8f;

#define DDIM 1024
#define HDIM 4096
#define NBLK 16
#define MTOK 32          // tokens per workgroup
#define KCH  512         // H-chunk width (2 expert blocks)
#define XSTR (DDIM + 8)  // LDS row stride (bf16 elems), +16B pad -> 2-way-optimal banks
#define HSTR (KCH + 8)

static __device__ __forceinline__ v8f wmma_bf16(bf16x16 a, bf16x16 b, v8f c) {
  // D(f32 16x16) = A(16x32 bf16) * B(32x16 bf16) + C
  return __builtin_amdgcn_wmma_f32_16x16x32_bf16(false, a, false, b, (short)0, c, false, false);
}

// Per-lane A/B fragment load: two 16-byte chunks, K {0..7,16..23} (lanes 0-15)
// or K {8..15,24..31} (lanes 16-31); caller bakes half8 into `base`.
static __device__ __forceinline__ bf16x16 load_frag(const unsigned short* base) {
  bf16x8 lo = *reinterpret_cast<const bf16x8*>(base);
  bf16x8 hi = *reinterpret_cast<const bf16x8*>(base + 16);
  bf16x16 r;
  #pragma unroll
  for (int i = 0; i < 8; ++i) { r[i] = lo[i]; r[i + 8] = hi[i]; }
  return r;
}

// ---------------------------------------------------------------------------
// Kernel 1: f32 -> bf16 weight conversion (one elem of w1 and w2 per thread)
// ---------------------------------------------------------------------------
__global__ __launch_bounds__(256) void cvt_kernel(
    const float* __restrict__ w1, const float* __restrict__ w2,
    unsigned short* __restrict__ w1b, unsigned short* __restrict__ w2b)
{
  int i = blockIdx.x * 256 + threadIdx.x;           // grid sized exactly: H*D / 256
  union { __bf16 b; unsigned short u; } c1, c2;
  c1.b = (__bf16)w1[i];
  c2.b = (__bf16)w2[i];
  w1b[i] = c1.u;
  w2b[i] = c2.u;
}

// ---------------------------------------------------------------------------
// Kernel 2: router. One wave per token. Softmax is monotonic -> top-4 of
// logits == top-4 of probs. Emits 16-bit expert bitmask per token.
// ---------------------------------------------------------------------------
__global__ __launch_bounds__(256) void router_kernel(
    const float* __restrict__ x, const float* __restrict__ wr,
    const float* __restrict__ br, unsigned int* __restrict__ maskOut, int N)
{
  int tok  = blockIdx.x * (blockDim.x >> 5) + (threadIdx.x >> 5);
  int lane = threadIdx.x & 31;
  if (tok >= N) return;
  const float* xr = x + (size_t)tok * DDIM;

  float acc[NBLK];
  #pragma unroll
  for (int e = 0; e < NBLK; ++e) acc[e] = 0.f;

  for (int i = lane; i < DDIM; i += 32) {           // coalesced x reads
    float xv = xr[i];
    const float4* w4 = reinterpret_cast<const float4*>(wr + (size_t)i * NBLK);
    #pragma unroll
    for (int q = 0; q < 4; ++q) {
      float4 w = w4[q];
      acc[q * 4 + 0] += xv * w.x;
      acc[q * 4 + 1] += xv * w.y;
      acc[q * 4 + 2] += xv * w.z;
      acc[q * 4 + 3] += xv * w.w;
    }
  }
  #pragma unroll
  for (int e = 0; e < NBLK; ++e) {
    float v = acc[e];
    for (int m = 16; m >= 1; m >>= 1) v += __shfl_xor(v, m, 32);
    acc[e] = v + br[e];
  }
  if (lane == 0) {
    unsigned sel = 0;
    #pragma unroll
    for (int k = 0; k < 4; ++k) {                   // 4x argmax, earliest index wins ties
      int bi = 0; float bv = -FLT_MAX;
      #pragma unroll
      for (int e = 0; e < NBLK; ++e) {
        bool taken = (sel >> e) & 1u;
        if (!taken && acc[e] > bv) { bv = acc[e]; bi = e; }
      }
      sel |= 1u << bi;
    }
    maskOut[tok] = sel;
  }
}

// ---------------------------------------------------------------------------
// Kernel 3: block-sparse FFN as masked dense, WMMA bf16, f32 accumulate.
// 8 waves / WG, 32 tokens / WG (two 16-row M-tiles per wave).
// ---------------------------------------------------------------------------
__global__ __launch_bounds__(256) void ffn_kernel(
    const float* __restrict__ x,
    const float* __restrict__ b1, const float* __restrict__ b2,
    const unsigned short* __restrict__ w1b, const unsigned short* __restrict__ w2b,
    const unsigned int* __restrict__ mask,
    float* __restrict__ out)
{
  extern __shared__ unsigned short smem[];
  unsigned short* Xs = smem;                         // [32][XSTR] bf16
  unsigned short* Hs = smem + MTOK * XSTR;           // [32][HSTR] bf16
  unsigned int*   Ms = reinterpret_cast<unsigned int*>(smem + MTOK * (XSTR + HSTR));

  const int tid    = threadIdx.x;
  const int wave   = tid >> 5;
  const int lane   = tid & 31;
  const int lane15 = lane & 15;
  const int half8  = (lane >> 4) << 3;               // 0 (lanes 0-15) / 8 (lanes 16-31)
  const size_t tok0 = (size_t)blockIdx.x * MTOK;

  // --- stage X tile (f32 -> bf16) + routing masks into LDS ---
  for (int i = tid; i < MTOK * (DDIM / 4); i += 256) {
    int row = i >> 8;                                // DDIM/4 = 256 float4 per row
    int c4  = i & 255;
    float4 v = reinterpret_cast<const float4*>(x + (tok0 + row) * DDIM)[c4];
    unsigned short* p = &Xs[row * XSTR + c4 * 4];
    union { __bf16 b; unsigned short u; } t0, t1, t2, t3;
    t0.b = (__bf16)v.x; t1.b = (__bf16)v.y; t2.b = (__bf16)v.z; t3.b = (__bf16)v.w;
    p[0] = t0.u; p[1] = t1.u; p[2] = t2.u; p[3] = t3.u;
  }
  if (tid < MTOK) Ms[tid] = mask[tok0 + tid];
  __syncthreads();

  v8f accY[16];                                      // [mt][8 N-tiles], 128 cols of D per wave
  #pragma unroll
  for (int i = 0; i < 16; ++i) accY[i] = (v8f)(0.0f);

  const int cbCol = wave * 128;                      // phase-B column base (d)

  for (int c = 0; c < HDIM / KCH; ++c) {
    const int ca = c * KCH + wave * 64;              // phase-A column base (h unit)
    const int e  = ca >> 8;                          // expert block of this wave's 64 cols

    // per-lane routing bits for the 8 rows this lane half touches, per M-tile
    unsigned mbits0 = 0, mbits1 = 0;
    #pragma unroll
    for (int j = 0; j < 8; ++j) {
      mbits0 |= ((Ms[half8 + j]      >> e) & 1u) << j;
      mbits1 |= ((Ms[16 + half8 + j] >> e) & 1u) << j;
    }

    // ---- phase A: H[32, 64 cols] = X @ W1^T ----
    v8f acc1[8];
    #pragma unroll
    for (int i = 0; i < 8; ++i) acc1[i] = (v8f)(0.0f);

    for (int kk = 0; kk < DDIM; kk += 32) {
      bf16x16 a0 = load_frag(&Xs[(lane15)      * XSTR + kk + half8]);
      bf16x16 a1 = load_frag(&Xs[(16 + lane15) * XSTR + kk + half8]);
      #pragma unroll
      for (int t = 0; t < 4; ++t) {
        const unsigned short* wp =
            w1b + (size_t)(ca + t * 16 + lane15) * DDIM + kk + half8;
        bf16x16 bfr = load_frag(wp);
        acc1[t]     = wmma_bf16(a0, bfr, acc1[t]);
        acc1[4 + t] = wmma_bf16(a1, bfr, acc1[4 + t]);
      }
    }

    __syncthreads();                                 // previous Hs fully consumed

    // ---- relu + bias + routing mask, write bf16 H to LDS ----
    #pragma unroll
    for (int mt = 0; mt < 2; ++mt) {
      unsigned mb = mt ? mbits1 : mbits0;
      #pragma unroll
      for (int t = 0; t < 4; ++t) {
        float bias = b1[ca + t * 16 + lane15];
        v8f a = acc1[mt * 4 + t];
        #pragma unroll
        for (int j = 0; j < 8; ++j) {
          float v = a[j] + bias;
          v = v > 0.f ? v : 0.f;
          if (!((mb >> j) & 1u)) v = 0.f;
          int row = mt * 16 + half8 + j;
          int col = wave * 64 + t * 16 + lane15;     // chunk-local column
          union { __bf16 b; unsigned short u; } cv; cv.b = (__bf16)v;
          Hs[row * HSTR + col] = cv.u;
        }
      }
    }
    __syncthreads();

    // ---- phase B: Y += H_chunk @ W2[:, chunk cols]^T ----
    for (int kk = 0; kk < KCH; kk += 32) {
      bf16x16 a0 = load_frag(&Hs[(lane15)      * HSTR + kk + half8]);
      bf16x16 a1 = load_frag(&Hs[(16 + lane15) * HSTR + kk + half8]);
      const int kglob = c * KCH + kk;
      #pragma unroll
      for (int t = 0; t < 8; ++t) {
        const unsigned short* wp =
            w2b + (size_t)(cbCol + t * 16 + lane15) * HDIM + kglob + half8;
        bf16x16 bfr = load_frag(wp);
        accY[t]     = wmma_bf16(a0, bfr, accY[t]);
        accY[8 + t] = wmma_bf16(a1, bfr, accY[8 + t]);
      }
    }
  }

  // ---- epilogue: + b2, store f32 ----
  #pragma unroll
  for (int mt = 0; mt < 2; ++mt) {
    #pragma unroll
    for (int t = 0; t < 8; ++t) {
      int col = cbCol + t * 16 + lane15;
      float bias = b2[col];
      v8f a = accY[mt * 8 + t];
      #pragma unroll
      for (int j = 0; j < 8; ++j) {
        int row = mt * 16 + half8 + j;
        out[(tok0 + row) * DDIM + col] = a[j] + bias;
      }
    }
  }
}

// ---------------------------------------------------------------------------
extern "C" void kernel_launch(void* const* d_in, const int* in_sizes, int n_in,
                              void* d_out, int out_size, void* d_ws, size_t ws_size,
                              hipStream_t stream) {
  const float* x  = (const float*)d_in[0];
  const float* wr = (const float*)d_in[1];
  const float* br = (const float*)d_in[2];
  const float* w1 = (const float*)d_in[3];
  const float* b1 = (const float*)d_in[4];
  const float* w2 = (const float*)d_in[5];
  const float* b2 = (const float*)d_in[6];
  float* out = (float*)d_out;

  const int N = in_sizes[0] / DDIM;                  // 16384 tokens
  const size_t WELEMS = (size_t)HDIM * DDIM;         // 4M elems each

  unsigned short* w1b  = (unsigned short*)d_ws;
  unsigned short* w2b  = w1b + WELEMS;
  unsigned int*   mask = (unsigned int*)(w2b + WELEMS);

  // 1) weights -> bf16
  cvt_kernel<<<(int)(WELEMS / 256), 256, 0, stream>>>(w1, w2, w1b, w2b);

  // 2) router: one wave per token, 8 waves per block
  router_kernel<<<(N + 7) / 8, 256, 0, stream>>>(x, wr, br, mask, N);

  // 3) FFN: 32 tokens per WG
  const int lds_bytes = (MTOK * (XSTR + HSTR)) * 2 + MTOK * 4;  // ~99.5 KB
  ffn_kernel<<<N / MTOK, 256, lds_bytes, stream>>>(x, b1, b2, w1b, w2b, mask, out);
}
